// LSTM_CRF_1391569404391
// MI455X (gfx1250) — matrix operand order repaired and checked by reading, not compiled, not run on hardware
//
#include <hip/hip_runtime.h>
#include <stdint.h>

// ---- problem dims ----
#define Lseq 512
#define Bsz  256
#define Edim 50
#define HD   150
#define G4   600     // 4*HD gate width
#define NPAD 640     // gate width padded to 40 x 16 tiles
#define KE   64      // E padded for K=32 WMMA
#define KH   160     // HD padded for K=32 WMMA
#define KCAT 320     // 2*HD padded
#define Tt   9

typedef __attribute__((ext_vector_type(16))) __bf16 v16bf;
typedef __attribute__((ext_vector_type(8)))  float  v8f;

struct U32x8 { uint4 lo, hi; };

__device__ __forceinline__ v16bf frag_from(uint4 lo, uint4 hi) {
  U32x8 u{lo, hi};
  return __builtin_bit_cast(v16bf, u);
}

__device__ __forceinline__ unsigned short f2bf(float x) {
  unsigned u = __builtin_bit_cast(unsigned, x);
  unsigned r = u + 0x7FFFu + ((u >> 16) & 1u);   // round-to-nearest-even
  return (unsigned short)(r >> 16);
}

__device__ __forceinline__ float sigm(float x) { return 1.0f / (1.0f + __expf(-x)); }

// ---------------------------------------------------------------------------
// Prep: convert weights to zero-padded bf16 fragment-friendly layouts.
//   wih_bf : [2][NPAD][KE], whh_bf : [2][NPAD][KH], wtag_bf : [16][KCAT]
// ---------------------------------------------------------------------------
__global__ void prep_weights(const float* __restrict__ wihf, const float* __restrict__ wihb,
                             const float* __restrict__ whhf, const float* __restrict__ whhb,
                             const float* __restrict__ wtag,
                             unsigned short* __restrict__ wih_bf,
                             unsigned short* __restrict__ whh_bf,
                             unsigned short* __restrict__ wtag_bf) {
  int i = blockIdx.x * 256 + threadIdx.x;
  const int nIH = 2 * NPAD * KE;     // 81920
  const int nHH = 2 * NPAD * KH;     // 204800
  const int nTG = 16 * KCAT;         // 5120
  if (i < nIH) {
    int dir = i / (NPAD * KE); int r = i % (NPAD * KE);
    int n = r / KE, k = r % KE;
    float v = 0.f;
    if (n < G4 && k < Edim) v = (dir ? wihb : wihf)[n * Edim + k];
    wih_bf[i] = f2bf(v);
  } else if (i < nIH + nHH) {
    int j = i - nIH;
    int dir = j / (NPAD * KH); int r = j % (NPAD * KH);
    int n = r / KH, k = r % KH;
    float v = 0.f;
    if (n < G4 && k < HD) v = (dir ? whhb : whhf)[n * HD + k];
    whh_bf[j] = f2bf(v);
  } else if (i < nIH + nHH + nTG) {
    int j = i - nIH - nHH;
    int n = j / KCAT, k = j % KCAT;
    float v = 0.f;
    if (n < Tt && k < 2 * HD) v = wtag[n * (2 * HD) + k];
    wtag_bf[j] = f2bf(v);
  }
}

// Zero the K-pad columns (300..319) of the bf16 lstm concat buffer.
__global__ void pad_cat(unsigned short* __restrict__ cat) {
  int i = blockIdx.x * 256 + threadIdx.x;          // exactly 131072*20 threads
  int row = i / 20, j = i - row * 20;
  cat[(size_t)row * KCAT + 300 + j] = 0;
}

// ---------------------------------------------------------------------------
// Gate precompute: G[dir][l*B+b][0..599] = x @ W_ih^T + bias   (bf16 WMMA)
// One block per 16-row tile of (l,b); 8 waves x 10 (dir,n-tile) jobs.
// ---------------------------------------------------------------------------
__global__ __launch_bounds__(256) void gates_kernel(
    const int* __restrict__ sentences, const float* __restrict__ embedding,
    const unsigned short* __restrict__ wih_bf,
    const float* __restrict__ bfw, const float* __restrict__ bbw,
    float* __restrict__ Gf, float* __restrict__ Gb) {
  __shared__ __align__(16) unsigned short x_bf[16 * KE];
  const int tid = threadIdx.x;
  const int R0  = blockIdx.x * 16;
  const int l   = R0 >> 8;          // row = l*256 + b
  const int b0  = R0 & 255;
  {
    int m  = tid >> 4;
    int k0 = (tid & 15) * 4;
    int sid = sentences[(b0 + m) * Lseq + l];
    const float* erow = embedding + (size_t)sid * Edim;
#pragma unroll
    for (int j = 0; j < 4; ++j) {
      int k = k0 + j;
      x_bf[m * KE + k] = (k < Edim) ? f2bf(erow[k]) : (unsigned short)0;
    }
  }
  __syncthreads();

  const int lane = tid & 31;
  const int wv   = tid >> 5;
  const int mA   = lane & 15;
  const int kbA  = (lane < 16) ? 0 : 8;
  const int kbB  = (lane < 16) ? 0 : 8;
  const int mhi  = (lane >> 4) << 3;

  for (int job = wv; job < 80; job += 8) {       // 2 dirs x 40 n-tiles
    const int dir = job / 40;
    const int n0  = (job % 40) * 16;
    const int nCol = n0 + mA;
    float bias = 0.f;
    if (nCol < G4) bias = dir ? bbw[nCol] : bfw[nCol];
    v8f acc;
#pragma unroll
    for (int r = 0; r < 8; ++r) acc[r] = bias;
#pragma unroll
    for (int kk = 0; kk < 2; ++kk) {
      const int K0 = kk * 32;
      const unsigned short* ar = x_bf + mA * KE + K0 + kbA;
      v16bf A = frag_from(*(const uint4*)ar, *(const uint4*)(ar + 16));
      const unsigned short* br = wih_bf + ((size_t)(dir * NPAD + nCol)) * KE + K0 + kbB;
      v16bf Bm = frag_from(*(const uint4*)br, *(const uint4*)(br + 16));
      acc = __builtin_amdgcn_wmma_f32_16x16x32_bf16(false, A, false, Bm, (short)0, acc, false, false);
    }
    if (nCol < G4) {
      float* Gd = dir ? Gb : Gf;
#pragma unroll
      for (int r = 0; r < 8; ++r)
        Gd[(size_t)(R0 + r + mhi) * G4 + nCol] = acc[r];
    }
  }
}

// ---------------------------------------------------------------------------
// Recurrent BiLSTM: one block per (dir, 16 batch rows). W_hh bf16 in LDS.
// Per step: 25 bf16 WMMAs/wave from LDS, then elementwise gates in f32.
// Writes masked h (bf16) straight into concat buffer cat[row][dir*150+hd].
// ---------------------------------------------------------------------------
__global__ __launch_bounds__(256) void lstm_kernel(
    const float* __restrict__ Gf, const float* __restrict__ Gb,
    const unsigned short* __restrict__ whh_bf,
    const int* __restrict__ lengths, unsigned short* __restrict__ cat) {
  extern __shared__ __align__(16) char smem[];
  unsigned short* W_lds = (unsigned short*)smem;                                   // 640*160 bf16
  float*          g_lds = (float*)(smem + (size_t)NPAD * KH * 2);                  // 16*640 f32
  unsigned short* h_lds = (unsigned short*)(smem + (size_t)NPAD*KH*2 + 16*NPAD*4); // 16*160 bf16
  float*          c_lds = (float*)(smem + (size_t)NPAD*KH*2 + 16*NPAD*4 + 16*KH*2);// 16*152 f32
  __shared__ int len_s[16];

  const int tid = threadIdx.x;
  const int dir = blockIdx.x >> 4;
  const int b0  = (blockIdx.x & 15) * 16;
  const float* Gd = dir ? Gb : Gf;

  {
    const unsigned int* src = (const unsigned int*)(whh_bf + (size_t)dir * NPAD * KH);
    unsigned int* dst = (unsigned int*)W_lds;
    for (int i = tid; i < NPAD * KH / 2; i += 256) dst[i] = src[i];
    unsigned int* hz = (unsigned int*)h_lds;
    for (int i = tid; i < 16 * KH / 2; i += 256) hz[i] = 0;         // h=0 incl. K-pad
    for (int i = tid; i < 16 * 152; i += 256) c_lds[i] = 0.f;
    if (tid < 16) len_s[tid] = lengths[b0 + tid];
  }
  __syncthreads();

  const int lane = tid & 31;
  const int wv   = tid >> 5;
  const int mA   = lane & 15;
  const int kbA  = (lane < 16) ? 0 : 8;
  const int kbB  = (lane < 16) ? 0 : 8;
  const int mhi  = (lane >> 4) << 3;

  for (int it = 0; it < Lseq; ++it) {
    const int t = dir ? (Lseq - 1 - it) : it;
    // ---- phase 1: g_lds = h @ W_hh^T ----
    v8f acc[5];
#pragma unroll
    for (int j = 0; j < 5; ++j)
#pragma unroll
      for (int r = 0; r < 8; ++r) acc[j][r] = 0.f;
#pragma unroll
    for (int kk = 0; kk < 5; ++kk) {
      const int K0 = kk * 32;
      const unsigned short* ar = h_lds + mA * KH + K0 + kbA;
      v16bf A = frag_from(*(const uint4*)ar, *(const uint4*)(ar + 16));
#pragma unroll
      for (int j = 0; j < 5; ++j) {
        const int nCol = (wv * 5 + j) * 16 + mA;
        const unsigned short* br = W_lds + nCol * KH + K0 + kbB;
        v16bf Bm = frag_from(*(const uint4*)br, *(const uint4*)(br + 16));
        acc[j] = __builtin_amdgcn_wmma_f32_16x16x32_bf16(false, A, false, Bm, (short)0, acc[j], false, false);
      }
    }
#pragma unroll
    for (int j = 0; j < 5; ++j) {
      const int nCol = (wv * 5 + j) * 16 + mA;
#pragma unroll
      for (int r = 0; r < 8; ++r)
        g_lds[(r + mhi) * NPAD + nCol] = acc[j][r];
    }
    __syncthreads();
    // ---- phase 2: gates + state update ----
    const int tn = dir ? (t - 1) : (t + 1);
    for (int i = tid; i < 16 * HD; i += 256) {
      const int m  = i / HD;
      const int hd = i - m * HD;
      const size_t Rrow = ((size_t)t << 8) + b0 + m;
      const float* Gr = Gd + Rrow * G4;
      float gi = g_lds[m * NPAD + hd]          + Gr[hd];
      float gf = g_lds[m * NPAD + HD + hd]     + Gr[HD + hd];
      float gg = g_lds[m * NPAD + 2 * HD + hd] + Gr[2 * HD + hd];
      float go = g_lds[m * NPAD + 3 * HD + hd] + Gr[3 * HD + hd];
      float c_old = c_lds[m * 152 + hd];
      float cn = sigm(gf) * c_old + sigm(gi) * tanhf(gg);
      float hn = sigm(go) * tanhf(cn);
      const bool mt = t < len_s[m];
      if (mt) {
        c_lds[m * 152 + hd] = cn;
        h_lds[m * KH + hd]  = f2bf(hn);
      }
      cat[Rrow * KCAT + dir * HD + hd] = f2bf(mt ? hn : 0.f);
      if (tn >= 0 && tn < Lseq) {   // prefetch next step's gate inputs
        const float* Gn = Gd + (((size_t)tn << 8) + b0 + m) * G4;
        __builtin_prefetch(Gn + hd, 0, 1);
        __builtin_prefetch(Gn + HD + hd, 0, 1);
        __builtin_prefetch(Gn + 2 * HD + hd, 0, 1);
        __builtin_prefetch(Gn + 3 * HD + hd, 0, 1);
      }
    }
    __syncthreads();
  }
}

// ---------------------------------------------------------------------------
// Emissions: emis[l*B+b][0..15] = cat @ w_tag^T + b_tag (cols >= 9 are zero)
// One wave per 16-row tile; 10 bf16 WMMAs.
// ---------------------------------------------------------------------------
__global__ __launch_bounds__(256) void emis_kernel(
    const unsigned short* __restrict__ cat, const unsigned short* __restrict__ wtag_bf,
    const float* __restrict__ b_tag, float* __restrict__ emis) {
  const int tid  = threadIdx.x;
  const int lane = tid & 31;
  const int wv   = tid >> 5;
  const int R0   = (blockIdx.x * 8 + wv) * 16;
  const int mA   = lane & 15;
  const int kbA  = (lane < 16) ? 0 : 8;
  const int kbB  = (lane < 16) ? 0 : 8;
  float bias = (mA < Tt) ? b_tag[mA] : 0.f;
  v8f acc;
#pragma unroll
  for (int r = 0; r < 8; ++r) acc[r] = bias;
#pragma unroll
  for (int kk = 0; kk < 10; ++kk) {
    const int K0 = kk * 32;
    const unsigned short* ar = cat + ((size_t)(R0 + mA)) * KCAT + K0 + kbA;
    v16bf A = frag_from(*(const uint4*)ar, *(const uint4*)(ar + 16));
    const unsigned short* br = wtag_bf + mA * KCAT + K0 + kbB;
    v16bf Bm = frag_from(*(const uint4*)br, *(const uint4*)(br + 16));
    acc = __builtin_amdgcn_wmma_f32_16x16x32_bf16(false, A, false, Bm, (short)0, acc, false, false);
  }
  const int mhi = (lane >> 4) << 3;
#pragma unroll
  for (int r = 0; r < 8; ++r)
    emis[(size_t)(R0 + r + mhi) * 16 + mA] = acc[r];
}

// ---------------------------------------------------------------------------
// CRF: one wave32 per batch element. Lanes 0..8 hold the 9 tags.
// ---------------------------------------------------------------------------
__global__ void crf_kernel(const float* __restrict__ emis, const int* __restrict__ targets,
                           const int* __restrict__ lengths,
                           const float* __restrict__ start_t, const float* __restrict__ end_t,
                           const float* __restrict__ trans, float* __restrict__ out) {
  const int b    = blockIdx.x;
  const int lane = threadIdx.x;
  const bool act = lane < Tt;
  const int len  = lengths[b];

  float trc[Tt];
#pragma unroll
  for (int t = 0; t < Tt; ++t) trc[t] = act ? trans[t * Tt + lane] : 0.f;

  float alpha = act ? (start_t[lane] + emis[(size_t)b * 16 + lane]) : -1e30f;

  int tag_prev = targets[b * Lseq];
  float num = start_t[tag_prev] + emis[(size_t)b * 16 + tag_prev];

  for (int t = 1; t < Lseq; ++t) {
    const float* em = emis + ((size_t)t * Bsz + b) * 16;
    const bool mt = t < len;
    float emv = act ? em[lane] : 0.f;
    float vv[Tt];
    float mx = -1e30f;
#pragma unroll
    for (int tt = 0; tt < Tt; ++tt) {
      float av = __shfl(alpha, tt, 32);
      float v  = av + trc[tt];
      vv[tt] = v;
      mx = fmaxf(mx, v);
    }
    float s = 0.f;
#pragma unroll
    for (int tt = 0; tt < Tt; ++tt) s += __expf(vv[tt] - mx);
    float anew = emv + mx + __logf(s);
    if (mt && act) alpha = anew;

    int tc = targets[b * Lseq + t];
    if (mt) num += trans[tag_prev * Tt + tc] + em[tc];
    tag_prev = tc;
  }

  float numerator = num + end_t[targets[b * Lseq + (len - 1)]];

  float v = act ? (alpha + end_t[lane]) : -1e30f;
  float mx = v;
#pragma unroll
  for (int off = 16; off >= 1; off >>= 1) mx = fmaxf(mx, __shfl_xor(mx, off, 32));
  float s = __expf(v - mx);
#pragma unroll
  for (int off = 16; off >= 1; off >>= 1) s += __shfl_xor(s, off, 32);
  float den = mx + __logf(s);

  if (lane == 0) atomicAdd(out, numerator - den);
}

// ---------------------------------------------------------------------------
extern "C" void kernel_launch(void* const* d_in, const int* in_sizes, int n_in,
                              void* d_out, int out_size, void* d_ws, size_t ws_size,
                              hipStream_t stream) {
  (void)in_sizes; (void)n_in; (void)out_size; (void)ws_size;
  const int*   sentences = (const int*)  d_in[0];
  const int*   lengths   = (const int*)  d_in[1];
  const int*   targets   = (const int*)  d_in[2];
  const float* embedding = (const float*)d_in[3];
  const float* w_ih_f    = (const float*)d_in[4];
  const float* w_hh_f    = (const float*)d_in[5];
  const float* b_f       = (const float*)d_in[6];
  const float* w_ih_b    = (const float*)d_in[7];
  const float* w_hh_b    = (const float*)d_in[8];
  const float* b_b       = (const float*)d_in[9];
  const float* w_tag     = (const float*)d_in[10];
  const float* b_tag     = (const float*)d_in[11];
  const float* start_t   = (const float*)d_in[12];
  const float* end_t     = (const float*)d_in[13];
  const float* trans     = (const float*)d_in[14];

  char* ws = (char*)d_ws;
  const size_t szG   = (size_t)Lseq * Bsz * G4 * sizeof(float);          // 300 MB
  const size_t szCat = (size_t)Lseq * Bsz * KCAT * sizeof(unsigned short);
  const size_t szEm  = (size_t)Lseq * Bsz * 16 * sizeof(float);
  float*          Gf      = (float*)ws;
  float*          Gb      = (float*)(ws + szG);
  unsigned short* cat     = (unsigned short*)(ws + 2 * szG);
  float*          emis    = (float*)(ws + 2 * szG + szCat);
  unsigned short* wih_bf  = (unsigned short*)(ws + 2 * szG + szCat + szEm);
  unsigned short* whh_bf  = wih_bf + 2 * NPAD * KE;
  unsigned short* wtag_bf = whh_bf + 2 * NPAD * KH;

  hipMemsetAsync(d_out, 0, sizeof(float), stream);

  prep_weights<<<1140, 256, 0, stream>>>(w_ih_f, w_ih_b, w_hh_f, w_hh_b, w_tag,
                                         wih_bf, whh_bf, wtag_bf);
  pad_cat<<<(Lseq * Bsz * 20) / 256, 256, 0, stream>>>(cat);
  gates_kernel<<<Lseq * Bsz / 16, 256, 0, stream>>>(sentences, embedding, wih_bf,
                                                    b_f, b_b, Gf, Gb);
  const size_t lds_bytes = (size_t)NPAD * KH * 2 + (size_t)16 * NPAD * 4
                         + (size_t)16 * KH * 2 + (size_t)16 * 152 * 4;   // 260,608 B
  lstm_kernel<<<32, 256, lds_bytes, stream>>>(Gf, Gb, whh_bf, lengths, cat);
  emis_kernel<<<Lseq * Bsz / 128, 256, 0, stream>>>(cat, wtag_bf, b_tag, emis);
  crf_kernel<<<Bsz, 32, 0, stream>>>(emis, targets, lengths, start_t, end_t, trans,
                                     (float*)d_out);
}
